// RaSTER_4552665334042
// MI455X (gfx1250) — matrix operand order repaired and checked by reading, not compiled, not run on hardware
//
#include <hip/hip_runtime.h>

// ---------------------------------------------------------------------------
// CDNA5 / gfx1250 implementation of the 2-layer equivariant graph attention
// reference. Dense node GEMMs run on v_wmma_f32_16x16x32_bf16 (f32 accum);
// edge work (radial MLP, attention logits, segment-sum scatters) is
// VALU + global_atomic_add_f32, relying on the 192MB L2 to absorb gathers.
// ---------------------------------------------------------------------------

#define NDIM   176           // node feature dim
#define DIMP   192           // NDIM padded to a multiple of 32 (WMMA K)
#define ATT    16
#define SH_HEADS 16
#define THH    4
#define NHEADS 20            // SH_HEADS + TH
#define VD     11            // DIM / SH_HEADS
#define NSPH   16
#define QKV_COLS 816         // 320 (q) + 320 (k) + 176 (v)
#define UPD_K  352           // 2*DIM, already a multiple of 32

typedef __bf16 bf16_t;
typedef __attribute__((ext_vector_type(16))) __bf16 v16bf;
typedef __attribute__((ext_vector_type(8)))  float  v8f;

__device__ __forceinline__ float parity_of(int s) {
  // l = [0,1,1,1,2,2,2,2,2,3,...,3]; parity = (-1)^l
  return ((s >= 1 && s <= 3) || s >= 9) ? -1.f : 1.f;
}
__device__ __forceinline__ int l_of(int s) {
  return (s < 1) ? 0 : (s < 4) ? 1 : (s < 9) ? 2 : 3;
}

// ---------------------------------------------------------------------------
// WMMA GEMM: C[M x Ncols] = A[M x Kp] (bf16) * Bt[Ncols x Kp]^T (bf16)
// Bt is stored output-column-major (row n of Bt = column n of B), so both
// fragments are two contiguous 128-bit loads per lane.
// A frag (16x32 bf16): lane = {row = lane&15, khalf = (lane>>4)*8},
//   v[0..7] = K[khalf..khalf+7], v[8..15] = K[khalf+16..khalf+23].
// B frag (32x16 bf16): lane = col (lane&15), khalf = (lane>>4)*8,
//   v[0..7] = K[khalf..khalf+7], v[8..15] = K[khalf+16..khalf+23] (same as A).
// C/D frag: v[r] = C[row r + 8*(lane>>4), col lane&15].
// ---------------------------------------------------------------------------
__global__ __launch_bounds__(128)
void wmma_gemm_bf16_kernel(const bf16_t* __restrict__ A,
                           const bf16_t* __restrict__ Bt,
                           float* __restrict__ C,
                           int M, int Ncols, int Kp)
{
  const int tm   = blockIdx.x;
  const int tn   = blockIdx.y * blockDim.y + threadIdx.y;
  if (tn * 16 >= Ncols) return;
  const int lane = threadIdx.x;
  const int r16  = lane & 15;
  const int half = lane >> 4;

  const bf16_t* ap = A  + (size_t)(tm * 16 + r16) * Kp + half * 8;
  const bf16_t* bp = Bt + (size_t)(tn * 16 + r16) * Kp + half * 8;

  v8f acc = {};
  for (int k = 0; k < Kp; k += 32) {
    union { uint4 u[2]; v16bf v; } a, b;
    a.u[0] = *(const uint4*)(ap + k);
    a.u[1] = *(const uint4*)(ap + k + 16);
    b.u[0] = *(const uint4*)(bp + k);
    b.u[1] = *(const uint4*)(bp + k + 16);
    acc = __builtin_amdgcn_wmma_f32_16x16x32_bf16(
        /*neg_a=*/false, a.v, /*neg_b=*/false, b.v,
        /*c_mod=*/(short)0, acc, /*reuse_a=*/false, /*reuse_b=*/false);
  }
#pragma unroll
  for (int r = 0; r < 8; ++r) {
    int grow = tm * 16 + r + 8 * half;
    if (grow < M)
      C[(size_t)grow * Ncols + tn * 16 + r16] = acc[r];
  }
}

// ---------------------------------------------------------------------------
// Weight staging: W[K x Ncols] (f32, row-major) -> Bt rows [rowOff, rowOff+Ncols)
// each of length Kp bf16, transposed + zero padded in K.
// ---------------------------------------------------------------------------
__global__ void conv_weight_kernel(const float* __restrict__ W,
                                   bf16_t* __restrict__ Bt,
                                   int K, int Ncols, int Kp, int rowOff)
{
  size_t idx = (size_t)blockIdx.x * blockDim.x + threadIdx.x;
  if (idx >= (size_t)Ncols * Kp) return;
  int n = (int)(idx / Kp), k = (int)(idx % Kp);
  float v = (k < K) ? W[(size_t)k * Ncols + n] : 0.f;
  Bt[(size_t)(rowOff + n) * Kp + k] = (bf16_t)v;
}

__global__ void fill0_kernel(float* __restrict__ p, size_t n)
{
  size_t i = (size_t)blockIdx.x * blockDim.x + threadIdx.x;
  if (i < n) p[i] = 0.f;
}

// xi_pre = z_table[species] @ w_species  (ZDIM=16 inner product)
__global__ void embed_kernel(const int* __restrict__ sp,
                             const float* __restrict__ zt,
                             const float* __restrict__ wsp,
                             float* __restrict__ out, int Nn)
{
  size_t idx = (size_t)blockIdx.x * blockDim.x + threadIdx.x;
  if (idx >= (size_t)Nn * NDIM) return;
  int n = (int)(idx / NDIM), c = (int)(idx % NDIM);
  int z = sp[n];
  float acc = 0.f;
#pragma unroll
  for (int i = 0; i < 16; ++i) acc += zt[z * 16 + i] * wsp[i * NDIM + c];
  out[idx] = acc;
}

// ---------------------------------------------------------------------------
// Wave-per-row LayerNorm: out = LN(src [+ resid] [+ bias]); also writes the
// bf16 K-padded copy used as the next GEMM's A matrix.
// ---------------------------------------------------------------------------
__global__ __launch_bounds__(256)
void ln_kernel(const float* __restrict__ src,
               const float* __restrict__ resid,   // nullable
               const float* __restrict__ bias,    // nullable
               float* __restrict__ out_f32,
               bf16_t* __restrict__ out_b16,      // nullable, stride DIMP
               int Nrows)
{
  int row = blockIdx.x * blockDim.y + threadIdx.y;
  if (row >= Nrows) return;
  int lane = threadIdx.x;
  float tv[6];
  float s = 0.f;
#pragma unroll
  for (int j = 0; j < 6; ++j) {
    int c = lane + 32 * j;
    float t = 0.f;
    if (c < NDIM) {
      t = src[(size_t)row * NDIM + c];
      if (resid) t += resid[(size_t)row * NDIM + c];
      if (bias)  t += bias[c];
    }
    tv[j] = t;
    s += t;
  }
#pragma unroll
  for (int o = 16; o > 0; o >>= 1) s += __shfl_xor(s, o, 32);
  float mu = s * (1.f / (float)NDIM);
  float var = 0.f;
#pragma unroll
  for (int j = 0; j < 6; ++j) {
    int c = lane + 32 * j;
    if (c < NDIM) { float d = tv[j] - mu; var += d * d; }
  }
#pragma unroll
  for (int o = 16; o > 0; o >>= 1) var += __shfl_xor(var, o, 32);
  float rs = rsqrtf(1e-6f + var * (1.f / (float)NDIM));
#pragma unroll
  for (int j = 0; j < 6; ++j) {
    int c = lane + 32 * j;
    float o = (c < NDIM) ? (tv[j] - mu) * rs : 0.f;
    if (c < NDIM) out_f32[(size_t)row * NDIM + c] = o;
    if (out_b16 && c < DIMP) out_b16[(size_t)row * DIMP + c] = (bf16_t)o;
  }
}

// ---------------------------------------------------------------------------
// Per-edge geometry: spherical harmonics Y (l<=3) and Gaussian radial basis.
// ---------------------------------------------------------------------------
__global__ void edge_geom_kernel(const float* __restrict__ dist,
                                 const float* __restrict__ vec,
                                 float* __restrict__ Yij,
                                 float* __restrict__ rad, int E_)
{
  int e = blockIdx.x * blockDim.x + threadIdx.x;
  if (e >= E_) return;
  float d = dist[e];
  float inv = 1.f / d;
  float x = vec[e * 3 + 0] * inv, y = vec[e * 3 + 1] * inv, z = vec[e * 3 + 2] * inv;
  const float s3 = 1.73205081f, s15 = 3.87298335f;
  const float c1 = 0.79056942f, c2 = 0.61237244f;
  float x2 = x * x, y2 = y * y, z2 = z * z;
  float Y[16] = {1.f, x, y, z,
                 s3 * x * y, s3 * y * z, 0.5f * (3.f * z2 - 1.f), s3 * x * z,
                 0.5f * s3 * (x2 - y2),
                 c1 * y * (3.f * x2 - y2), s15 * x * y * z,
                 c2 * y * (5.f * z2 - 1.f), 0.5f * z * (5.f * z2 - 3.f),
                 c2 * x * (5.f * z2 - 1.f), 0.5f * s15 * z * (x2 - y2),
                 c1 * x * (x2 - 3.f * y2)};
#pragma unroll
  for (int i = 0; i < 16; ++i) Yij[(size_t)e * 16 + i] = Y[i];
  const float sigma = (5.0f - 0.8f) / 15.f;
#pragma unroll
  for (int i = 0; i < 16; ++i) {
    float c = 0.8f + (float)i * sigma;
    float t = (d - c) / sigma;
    rad[(size_t)e * 16 + i] = __expf(-t * t);
  }
}

// ---------------------------------------------------------------------------
// Edge attention + segment-sum scatters. One thread per edge.
// IN_DIM = 16 (layer 0: radial only) or 32 (layer 1: radial ++ us from Vi).
// WRITE_VI=false on the last layer (Vi result is dead -> skip 64 atomics/edge).
// ---------------------------------------------------------------------------
template <int IN_DIM, bool WRITE_VI>
__global__ __launch_bounds__(256)
void edge_attn_kernel(const int* __restrict__ esrc, const int* __restrict__ edst,
                      const float* __restrict__ sw,
                      const float* __restrict__ Yij, const float* __restrict__ rad,
                      const float* __restrict__ qkv,      // N x 816: q|k|v
                      const float* __restrict__ ViOld,    // N x 64
                      float* __restrict__ ViNew,          // N x 64 (accum)
                      float* __restrict__ mi,             // N x 176 (accum)
                      const float* __restrict__ pw1, const float* __restrict__ pb1,
                      const float* __restrict__ pw2, const float* __restrict__ pb2,
                      int E_)
{
  int e = blockIdx.x * blockDim.x + threadIdx.x;
  if (e >= E_) return;
  const int s  = esrc[e];
  const int dd = edst[e];

  float ur[IN_DIM];
#pragma unroll
  for (int i = 0; i < 16; ++i) ur[i] = rad[(size_t)e * 16 + i];
  if (IN_DIM == 32) {
#pragma unroll
    for (int i = 16; i < IN_DIM; ++i) ur[i] = 0.f;
#pragma unroll
    for (int sidx = 0; sidx < 16; ++sidx) {
      float y = Yij[(size_t)e * 16 + sidx];
      float p = parity_of(sidx);
      int   l = l_of(sidx);
#pragma unroll
      for (int t = 0; t < THH; ++t) {
        float x2 = (ViOld[(size_t)dd * 64 + t * 16 + sidx] +
                    p * ViOld[(size_t)s * 64 + t * 16 + sidx]) * y;
        ur[16 + l * THH + t] += x2;
      }
    }
  }

  // radial MLP: swish(ur @ pw1 + pb1) @ pw2 + pb2  (weights are wave-uniform)
  float h1[32];
#pragma unroll
  for (int j = 0; j < 32; ++j) {
    float a = pb1[j];
#pragma unroll
    for (int i = 0; i < IN_DIM; ++i) a += ur[i] * pw1[i * 32 + j];
    h1[j] = a / (1.f + __expf(-a));
  }
  float w[ATT];
#pragma unroll
  for (int d = 0; d < ATT; ++d) {
    float a = pb2[d];
#pragma unroll
    for (int j = 0; j < 32; ++j) a += h1[j] * pw2[j * ATT + d];
    w[d] = a;
  }

  const float scale = sw[e] * 0.25f;             // switch * 1/sqrt(ATT)
  const float* qrow = qkv + (size_t)dd * QKV_COLS;
  const float* krow = qkv + (size_t)s  * QKV_COLS + 320;
  const float* vrow = qkv + (size_t)s  * QKV_COLS + 640;

  // scalar heads -> mi scatter
#pragma unroll 4
  for (int h = 0; h < SH_HEADS; ++h) {
    float a = 0.f;
#pragma unroll
    for (int d = 0; d < ATT; ++d) a += qrow[h * ATT + d] * krow[h * ATT + d] * w[d];
    a *= scale;
#pragma unroll
    for (int d = 0; d < VD; ++d)
      atomicAdd(&mi[(size_t)dd * NDIM + h * VD + d], a * vrow[h * VD + d]);
  }

  if (WRITE_VI) {
#pragma unroll
    for (int t = 0; t < THH; ++t) {
      int hh = SH_HEADS + t;
      float a = 0.f;
#pragma unroll
      for (int d = 0; d < ATT; ++d) a += qrow[hh * ATT + d] * krow[hh * ATT + d] * w[d];
      a *= scale;
#pragma unroll
      for (int sidx = 0; sidx < 16; ++sidx) {
        float p = parity_of(sidx);
        atomicAdd(&ViNew[(size_t)dd * 64 + t * 16 + sidx],
                  a * (Yij[(size_t)e * 16 + sidx] +
                       p * ViOld[(size_t)s * 64 + t * 16 + sidx]));
      }
    }
  }
}

// A_upd[n, 0:176] = xi, A_upd[n, 176:352] = mi  (bf16 staging for update GEMM)
__global__ void concat_bf16_kernel(const float* __restrict__ xi,
                                   const float* __restrict__ mi,
                                   bf16_t* __restrict__ aupd, int Nn)
{
  size_t idx = (size_t)blockIdx.x * blockDim.x + threadIdx.x;
  if (idx >= (size_t)Nn * UPD_K) return;
  int n = (int)(idx / UPD_K), c = (int)(idx % UPD_K);
  float v = (c < NDIM) ? xi[(size_t)n * NDIM + c]
                       : mi[(size_t)n * NDIM + (c - NDIM)];
  aupd[idx] = (bf16_t)v;
}

// ---------------------------------------------------------------------------
extern "C" void kernel_launch(void* const* d_in, const int* in_sizes, int n_in,
                              void* d_out, int out_size, void* d_ws, size_t ws_size,
                              hipStream_t stream) {
  const int    Nn   = in_sizes[0];
  const int    Ee   = in_sizes[1];
  const int*   species = (const int*)  d_in[0];
  const float* dist    = (const float*)d_in[1];
  const float* sw      = (const float*)d_in[2];
  const int*   esrc    = (const int*)  d_in[3];
  const int*   edst    = (const int*)  d_in[4];
  const float* vec     = (const float*)d_in[5];
  const float* ztab    = (const float*)d_in[6];
  const float* wsp     = (const float*)d_in[7];
  const float* pw1[2]  = {(const float*)d_in[8],  (const float*)d_in[17]};
  const float* pb1[2]  = {(const float*)d_in[9],  (const float*)d_in[18]};
  const float* pw2[2]  = {(const float*)d_in[10], (const float*)d_in[19]};
  const float* pb2[2]  = {(const float*)d_in[11], (const float*)d_in[20]};
  const float* wq[2]   = {(const float*)d_in[12], (const float*)d_in[21]};
  const float* wk[2]   = {(const float*)d_in[13], (const float*)d_in[22]};
  const float* wv[2]   = {(const float*)d_in[14], (const float*)d_in[23]};
  const float* uw[2]   = {(const float*)d_in[15], (const float*)d_in[24]};
  const float* ub[2]   = {(const float*)d_in[16], (const float*)d_in[25]};

  const int Mt   = (Nn + 15) / 16;   // 625 for N=10000
  const int Mpad = Mt * 16;

  // --- workspace carve (256B aligned) ---
  char*  base = (char*)d_ws;
  size_t off  = 0;
  auto carve = [&](size_t bytes) -> void* {
    void* r = base + off;
    off += (bytes + 255) & ~(size_t)255;
    return r;
  };
  float*  xi    = (float*) carve((size_t)Mpad * NDIM * 4);
  bf16_t* xib   = (bf16_t*)carve((size_t)Mpad * DIMP * 2);
  float*  qkv   = (float*) carve((size_t)Mpad * QKV_COLS * 4);
  float*  mi    = (float*) carve((size_t)Nn * NDIM * 4);
  float*  ViA   = (float*) carve((size_t)Nn * 64 * 4);
  float*  ViB   = (float*) carve((size_t)Nn * 64 * 4);
  float*  Yij   = (float*) carve((size_t)Ee * 16 * 4);
  float*  rad   = (float*) carve((size_t)Ee * 16 * 4);
  bf16_t* aupd  = (bf16_t*)carve((size_t)Mpad * UPD_K * 2);
  float*  uout  = (float*) carve((size_t)Nn * NDIM * 4);  // also embed temp
  bf16_t* wqkvT = (bf16_t*)carve((size_t)QKV_COLS * DIMP * 2);
  bf16_t* updT  = (bf16_t*)carve((size_t)NDIM * UPD_K * 2);
  (void)ws_size; (void)n_in;

  // --- embedding + first layernorm ---
  {
    size_t tot = (size_t)Nn * NDIM;
    embed_kernel<<<(unsigned)((tot + 255) / 256), 256, 0, stream>>>(
        species, ztab, wsp, uout, Nn);
    ln_kernel<<<(Nn + 7) / 8, dim3(32, 8), 0, stream>>>(
        uout, nullptr, nullptr, xi, xib, Nn);
  }

  // --- edge geometry (shared by both layers) ---
  edge_geom_kernel<<<(Ee + 255) / 256, 256, 0, stream>>>(dist, vec, Yij, rad, Ee);

  // --- zero Vi buffers ---
  {
    size_t nv = (size_t)Nn * 64;
    fill0_kernel<<<(unsigned)((nv + 255) / 256), 256, 0, stream>>>(ViA, nv);
    fill0_kernel<<<(unsigned)((nv + 255) / 256), 256, 0, stream>>>(ViB, nv);
  }

  for (int layer = 0; layer < 2; ++layer) {
    // stage fused [wq|wk|wv] transposed to bf16, K padded 176 -> 192
    {
      size_t t1 = (size_t)320 * DIMP, t2 = (size_t)NDIM * DIMP;
      conv_weight_kernel<<<(unsigned)((t1 + 255) / 256), 256, 0, stream>>>(
          wq[layer], wqkvT, NDIM, 320, DIMP, 0);
      conv_weight_kernel<<<(unsigned)((t1 + 255) / 256), 256, 0, stream>>>(
          wk[layer], wqkvT, NDIM, 320, DIMP, 320);
      conv_weight_kernel<<<(unsigned)((t2 + 255) / 256), 256, 0, stream>>>(
          wv[layer], wqkvT, NDIM, NDIM, DIMP, 640);
    }
    // qkv = xi @ [wq|wk|wv]  : 51 column tiles, 4 waves/block
    wmma_gemm_bf16_kernel<<<dim3(Mt, (QKV_COLS / 16 + 3) / 4), dim3(32, 4), 0, stream>>>(
        xib, wqkvT, qkv, Nn, QKV_COLS, DIMP);

    // zero mi accumulator
    {
      size_t nm = (size_t)Nn * NDIM;
      fill0_kernel<<<(unsigned)((nm + 255) / 256), 256, 0, stream>>>(mi, nm);
    }

    // edge attention + scatters
    if (layer == 0) {
      edge_attn_kernel<16, true><<<(Ee + 255) / 256, 256, 0, stream>>>(
          esrc, edst, sw, Yij, rad, qkv, ViA, ViB, mi,
          pw1[0], pb1[0], pw2[0], pb2[0], Ee);
    } else {
      // Vi after layer 0 lives in ViB; layer-1 Vi output is dead -> no scatter
      edge_attn_kernel<32, false><<<(Ee + 255) / 256, 256, 0, stream>>>(
          esrc, edst, sw, Yij, rad, qkv, ViB, ViA, mi,
          pw1[1], pb1[1], pw2[1], pb2[1], Ee);
    }

    // update GEMM: [xi | mi] (N x 352) @ uw (352 x 176)
    {
      size_t tc = (size_t)Nn * UPD_K;
      concat_bf16_kernel<<<(unsigned)((tc + 255) / 256), 256, 0, stream>>>(
          xi, mi, aupd, Nn);
      size_t tw = (size_t)NDIM * UPD_K;
      conv_weight_kernel<<<(unsigned)((tw + 255) / 256), 256, 0, stream>>>(
          uw[layer], updT, UPD_K, NDIM, UPD_K, 0);
      wmma_gemm_bf16_kernel<<<dim3(Mt, (NDIM / 16 + 3) / 4), dim3(32, 4), 0, stream>>>(
          aupd, updT, uout, Nn, NDIM, UPD_K);
    }

    // xi = LN(xi + uout + ub); final layer writes d_out directly
    float*  of32 = (layer == 1) ? (float*)d_out : xi;
    bf16_t* ob16 = (layer == 1) ? nullptr : xib;
    ln_kernel<<<(Nn + 7) / 8, dim3(32, 8), 0, stream>>>(
        uout, xi, ub[layer], of32, ob16, Nn);
  }
  (void)out_size;
}
